// HeteroGNN_25598005084518
// MI455X (gfx1250) — matrix operand order repaired and checked by reading, not compiled, not run on hardware
//
#include <hip/hip_runtime.h>
#include <hip/hip_bf16.h>

#define H 64
#define N_PERSON 200000
#define N_MOVIE 100000
#define E_ACTED 1600000
#define E_DIRECTED 200000
#define E_WWW 3200000

typedef __attribute__((ext_vector_type(16))) __bf16 v16bf;
typedef __attribute__((ext_vector_type(8)))  float  v8f;

#define WPITCH 72  // LDS row pitch (bf16 elems): 16 column-lanes -> distinct banks

// ---------------------------------------------------------------------------
// WMMA: D = A(16x32 bf16) * B(32x16 bf16) + C(16x16 f32)
// ---------------------------------------------------------------------------
__device__ __forceinline__ v8f wmma_bf16(v16bf a, v16bf b, v8f c) {
  return __builtin_amdgcn_wmma_f32_16x16x32_bf16(
      false, a, false, b, (short)0, c, false, false);
}

// A fragment from f32 aggregate, scaled by 1/cnt (mean).
// Layout (ISA 7.12.2): M = lane%16 ; elems 0..7 -> K=ks*32+8*(lane/16)+e,
// elems 8..15 -> +16.
__device__ __forceinline__ v16bf load_A_mean(const float* __restrict__ rowp,
                                             int ks, int half, float scale) {
  const float* p = rowp + ks * 32 + 8 * half;
  v16bf a;
#pragma unroll
  for (int e = 0; e < 8; ++e) a[e] = (__bf16)(p[e] * scale);
#pragma unroll
  for (int e = 0; e < 8; ++e) a[8 + e] = (__bf16)(p[16 + e] * scale);
  return a;
}

// A fragment straight from bf16 activations (no conversion needed).
__device__ __forceinline__ v16bf load_A_bf16(const __bf16* __restrict__ rowp,
                                             int ks, int half) {
  const __bf16* p = rowp + ks * 32 + 8 * half;
  v16bf a;
#pragma unroll
  for (int e = 0; e < 8; ++e) a[e] = p[e];
#pragma unroll
  for (int e = 0; e < 8; ++e) a[8 + e] = p[16 + e];
  return a;
}

// B fragment from LDS-staged bf16 weights. col N = lane%16,
// elems 0..7 -> K = ks*32 + 8*(lane/16) + e, elems 8..15 -> +16 ; B[k][n] = W[n][k] (out = x @ W^T).
__device__ __forceinline__ v16bf load_B_lds(const __bf16* s, int n, int ks,
                                            int half) {
  const __bf16* p = s + n * WPITCH + ks * 32 + 8 * half;
  v16bf b;
#pragma unroll
  for (int e = 0; e < 8; ++e) b[e] = p[e];
#pragma unroll
  for (int e = 0; e < 8; ++e) b[8 + e] = p[16 + e];
  return b;
}

// ---------------------------------------------------------------------------
// f32 -> bf16 activation conversion
// ---------------------------------------------------------------------------
__global__ void to_bf16_kernel(const float* __restrict__ in,
                               __bf16* __restrict__ out, int n) {
  int i = blockIdx.x * blockDim.x + threadIdx.x;
  if (i < n) out[i] = (__bf16)in[i];
}

// ---------------------------------------------------------------------------
// Edge scatter: one wave per edge, 2 features/thread, 128B coalesced bf16
// gather, f32 atomics into L2-resident aggregate.
// ---------------------------------------------------------------------------
__global__ void scatter_kernel(const __bf16* __restrict__ x,
                               const int* __restrict__ src,
                               const int* __restrict__ dst,
                               float* __restrict__ agg,
                               float* __restrict__ cnt,
                               int nedges, int do_cnt) {
  long long t = (long long)blockIdx.x * blockDim.x + threadIdx.x;
  int e = (int)(t >> 5);
  if (e >= nedges) return;
  int lane = (int)t & 31;
  int h2 = lane * 2;
  int s = src[e];
  int d = dst[e];
  const __bf16* xp = x + (long long)s * H + h2;
  float* ap = agg + (long long)d * H + h2;
  atomicAdd(ap,     (float)xp[0]);
  atomicAdd(ap + 1, (float)xp[1]);
  if (do_cnt && lane == 0) atomicAdd(&cnt[d], 1.0f);
}

// ---------------------------------------------------------------------------
// Movie GEMM: relu(mean_a@Wl_a^T + bl_a + mean_d@Wl_d^T + bl_d
//                  + x_m@(Wr_a+Wr_d)^T). One wave per 16-row tile, weights
// staged once per block in LDS (bf16, padded pitch). 24 WMMAs/tile.
// ---------------------------------------------------------------------------
template <bool OUT_BF16>
__global__ void movie_gemm_kernel(const float* __restrict__ agg_a,
                                  const float* __restrict__ cnt_a,
                                  const float* __restrict__ agg_d,
                                  const float* __restrict__ cnt_d,
                                  const __bf16* __restrict__ xm_bf,
                                  const float* __restrict__ Wl_a,
                                  const float* __restrict__ bl_a,
                                  const float* __restrict__ Wr_a,
                                  const float* __restrict__ Wl_d,
                                  const float* __restrict__ bl_d,
                                  const float* __restrict__ Wr_d,
                                  float* __restrict__ outf,
                                  __bf16* __restrict__ outb) {
  __shared__ __bf16 sWl_a[H * WPITCH];
  __shared__ __bf16 sWl_d[H * WPITCH];
  __shared__ __bf16 sWs[H * WPITCH];
  for (int idx = threadIdx.x; idx < H * H; idx += blockDim.x) {
    int n = idx >> 6, k = idx & 63;
    sWl_a[n * WPITCH + k] = (__bf16)Wl_a[idx];
    sWl_d[n * WPITCH + k] = (__bf16)Wl_d[idx];
    sWs[n * WPITCH + k]   = (__bf16)(Wr_a[idx] + Wr_d[idx]);
  }
  __syncthreads();

  const int ntiles = N_MOVIE / 16;
  int wave = threadIdx.x >> 5;
  int lane = threadIdx.x & 31;
  int tile = blockIdx.x * (blockDim.x >> 5) + wave;
  if (tile >= ntiles) return;  // wave-uniform: EXEC all-1s for WMMA
  int half = lane >> 4;
  int l16  = lane & 15;
  int row  = tile * 16 + l16;

  float inva = __builtin_amdgcn_rcpf(fmaxf(cnt_a[row], 1.0f));
  float invd = __builtin_amdgcn_rcpf(fmaxf(cnt_d[row], 1.0f));

  v8f acc[4] = {};
#pragma unroll
  for (int ks = 0; ks < 2; ++ks) {
    v16bf a_a = load_A_mean(agg_a + (long long)row * H, ks, half, inva);
    v16bf a_d = load_A_mean(agg_d + (long long)row * H, ks, half, invd);
    v16bf a_s = load_A_bf16(xm_bf + (long long)row * H, ks, half);
#pragma unroll
    for (int nt = 0; nt < 4; ++nt) {
      int n = nt * 16 + l16;
      acc[nt] = wmma_bf16(a_a, load_B_lds(sWl_a, n, ks, half), acc[nt]);
      acc[nt] = wmma_bf16(a_d, load_B_lds(sWl_d, n, ks, half), acc[nt]);
      acc[nt] = wmma_bf16(a_s, load_B_lds(sWs, n, ks, half), acc[nt]);
    }
  }
  // C/D layout: VGPR r -> M = r + 8*(lane/16) ; N = lane%16
#pragma unroll
  for (int nt = 0; nt < 4; ++nt) {
    float b = bl_a[nt * 16 + l16] + bl_d[nt * 16 + l16];
#pragma unroll
    for (int r = 0; r < 8; ++r) {
      float v = fmaxf(acc[nt][r] + b, 0.0f);
      long long o = (long long)(tile * 16 + 8 * half + r) * H + nt * 16 + l16;
      if (OUT_BF16) outb[o] = (__bf16)v;
      else          outf[o] = v;
    }
  }
}

// Person GEMM: relu(mean_w@Wl_w^T + bl_w + x_p@Wr_w^T) -> bf16 (feeds gather)
__global__ void person_gemm_kernel(const float* __restrict__ agg_w,
                                   const float* __restrict__ cnt_w,
                                   const __bf16* __restrict__ xp_bf,
                                   const float* __restrict__ Wl_w,
                                   const float* __restrict__ bl_w,
                                   const float* __restrict__ Wr_w,
                                   __bf16* __restrict__ out) {
  __shared__ __bf16 sWl[H * WPITCH];
  __shared__ __bf16 sWr[H * WPITCH];
  for (int idx = threadIdx.x; idx < H * H; idx += blockDim.x) {
    int n = idx >> 6, k = idx & 63;
    sWl[n * WPITCH + k] = (__bf16)Wl_w[idx];
    sWr[n * WPITCH + k] = (__bf16)Wr_w[idx];
  }
  __syncthreads();

  const int ntiles = N_PERSON / 16;
  int wave = threadIdx.x >> 5;
  int lane = threadIdx.x & 31;
  int tile = blockIdx.x * (blockDim.x >> 5) + wave;
  if (tile >= ntiles) return;
  int half = lane >> 4;
  int l16  = lane & 15;
  int row  = tile * 16 + l16;

  float invw = __builtin_amdgcn_rcpf(fmaxf(cnt_w[row], 1.0f));

  v8f acc[4] = {};
#pragma unroll
  for (int ks = 0; ks < 2; ++ks) {
    v16bf a_w = load_A_mean(agg_w + (long long)row * H, ks, half, invw);
    v16bf a_s = load_A_bf16(xp_bf + (long long)row * H, ks, half);
#pragma unroll
    for (int nt = 0; nt < 4; ++nt) {
      int n = nt * 16 + l16;
      acc[nt] = wmma_bf16(a_w, load_B_lds(sWl, n, ks, half), acc[nt]);
      acc[nt] = wmma_bf16(a_s, load_B_lds(sWr, n, ks, half), acc[nt]);
    }
  }
#pragma unroll
  for (int nt = 0; nt < 4; ++nt) {
    float b = bl_w[nt * 16 + l16];
#pragma unroll
    for (int r = 0; r < 8; ++r) {
      float v = fmaxf(acc[nt][r] + b, 0.0f);
      out[(long long)(tile * 16 + 8 * half + r) * H + nt * 16 + l16] =
          (__bf16)v;
    }
  }
}

// Final: out[i] = dot(x_m2[i], lin_W) + lin_b
__global__ void final_linear_kernel(const float* __restrict__ xm,
                                    const float* __restrict__ W,
                                    const float* __restrict__ b,
                                    float* __restrict__ out) {
  int i = blockIdx.x * blockDim.x + threadIdx.x;
  if (i >= N_MOVIE) return;
  const float* row = xm + (long long)i * H;
  float s = 0.0f;
#pragma unroll
  for (int h = 0; h < H; ++h) s += row[h] * W[h];
  out[i] = s + b[0];
}

// ---------------------------------------------------------------------------
extern "C" void kernel_launch(void* const* d_in, const int* in_sizes, int n_in,
                              void* d_out, int out_size, void* d_ws, size_t ws_size,
                              hipStream_t stream) {
  (void)in_sizes; (void)n_in; (void)out_size; (void)ws_size;
  const float* x_p   = (const float*)d_in[0];
  const float* x_m   = (const float*)d_in[1];
  const int*   a_src = (const int*)d_in[2];
  const int*   a_dst = (const int*)d_in[3];
  const int*   dd_src= (const int*)d_in[4];
  const int*   dd_dst= (const int*)d_in[5];
  const int*   w_src = (const int*)d_in[6];
  const int*   w_dst = (const int*)d_in[7];
  const float* Wl_a  = (const float*)d_in[8];
  const float* bl_a  = (const float*)d_in[9];
  const float* Wr_a  = (const float*)d_in[10];
  const float* Wl_d  = (const float*)d_in[11];
  const float* bl_d  = (const float*)d_in[12];
  const float* Wr_d  = (const float*)d_in[13];
  const float* Wl_w  = (const float*)d_in[14];
  const float* bl_w  = (const float*)d_in[15];
  const float* Wr_w  = (const float*)d_in[16];
  const float* linW  = (const float*)d_in[17];
  const float* linb  = (const float*)d_in[18];
  float* out = (float*)d_out;

  // Workspace: f32 region then bf16 region (~207 MB total).
  float* ws    = (float*)d_ws;
  float* agg_a = ws;                                   // N_MOVIE*H f32
  float* agg_d = agg_a + (size_t)N_MOVIE * H;          // N_MOVIE*H
  float* agg_w = agg_d + (size_t)N_MOVIE * H;          // N_PERSON*H
  float* cnt_a = agg_w + (size_t)N_PERSON * H;         // N_MOVIE
  float* cnt_d = cnt_a + N_MOVIE;                      // N_MOVIE
  float* cnt_w = cnt_d + N_MOVIE;                      // N_PERSON
  float* xm2   = cnt_w + N_PERSON;                     // N_MOVIE*H f32
  __bf16* xp0b = (__bf16*)(xm2 + (size_t)N_MOVIE * H); // N_PERSON*H bf16
  __bf16* xm0b = xp0b + (size_t)N_PERSON * H;          // N_MOVIE*H
  __bf16* xm1b = xm0b + (size_t)N_MOVIE * H;           // N_MOVIE*H
  __bf16* xp1b = xm1b + (size_t)N_MOVIE * H;           // N_PERSON*H

  dim3 blk(256);
  const int WPB = 8;
  dim3 gm((N_MOVIE / 16 + WPB - 1) / WPB);
  dim3 gp((N_PERSON / 16 + WPB - 1) / WPB);

  // Zero layer-0 accumulators + counts (one contiguous memset).
  size_t zero_floats = (size_t)2 * N_MOVIE * H + (size_t)N_PERSON * H
                     + 2 * (size_t)N_MOVIE + (size_t)N_PERSON;
  hipMemsetAsync(agg_a, 0, zero_floats * sizeof(float), stream);

  // Activations -> bf16 (halves all gather traffic).
  {
    int np = N_PERSON * H, nm = N_MOVIE * H;
    to_bf16_kernel<<<dim3((np + 255) / 256), blk, 0, stream>>>(x_p, xp0b, np);
    to_bf16_kernel<<<dim3((nm + 255) / 256), blk, 0, stream>>>(x_m, xm0b, nm);
  }

  // -------- Layer 0 --------
  {
    long long ta = (long long)E_ACTED * 32, td = (long long)E_DIRECTED * 32,
              tw = (long long)E_WWW * 32;
    scatter_kernel<<<dim3((unsigned)((ta + 255) / 256)), blk, 0, stream>>>(
        xp0b, a_src, a_dst, agg_a, cnt_a, E_ACTED, 1);
    scatter_kernel<<<dim3((unsigned)((td + 255) / 256)), blk, 0, stream>>>(
        xp0b, dd_src, dd_dst, agg_d, cnt_d, E_DIRECTED, 1);
    scatter_kernel<<<dim3((unsigned)((tw + 255) / 256)), blk, 0, stream>>>(
        xp0b, w_src, w_dst, agg_w, cnt_w, E_WWW, 1);
  }
  movie_gemm_kernel<true><<<gm, blk, 0, stream>>>(
      agg_a, cnt_a, agg_d, cnt_d, xm0b,
      Wl_a, bl_a, Wr_a, Wl_d, bl_d, Wr_d, (float*)nullptr, xm1b);
  person_gemm_kernel<<<gp, blk, 0, stream>>>(agg_w, cnt_w, xp0b,
      Wl_w, bl_w, Wr_w, xp1b);

  // -------- Layer 1 (x_p output dead -> skip www; reuse layer-0 counts) ----
  hipMemsetAsync(agg_a, 0, (size_t)2 * N_MOVIE * H * sizeof(float), stream);
  {
    long long ta = (long long)E_ACTED * 32, td = (long long)E_DIRECTED * 32;
    scatter_kernel<<<dim3((unsigned)((ta + 255) / 256)), blk, 0, stream>>>(
        xp1b, a_src, a_dst, agg_a, cnt_a, E_ACTED, 0);
    scatter_kernel<<<dim3((unsigned)((td + 255) / 256)), blk, 0, stream>>>(
        xp1b, dd_src, dd_dst, agg_d, cnt_d, E_DIRECTED, 0);
  }
  movie_gemm_kernel<false><<<gm, blk, 0, stream>>>(
      agg_a, cnt_a, agg_d, cnt_d, xm1b,
      Wl_a + H * H, bl_a + H, Wr_a + H * H,
      Wl_d + H * H, bl_d + H, Wr_d + H * H, xm2, (__bf16*)nullptr);

  // -------- Final linear --------
  final_linear_kernel<<<dim3((N_MOVIE + 255) / 256), blk, 0, stream>>>(
      xm2, linW, linb, out);
}
